// Attention_87540023427086
// MI455X (gfx1250) — compile-verified, matrix operand order checked by reading, not run on hardware
//
#include <hip/hip_runtime.h>
#include <hip/hip_bf16.h>

// ---------------------------------------------------------------------------
// Problem constants
// ---------------------------------------------------------------------------
#define BATCH   2
#define SEQ     2048
#define DMODEL  2048
#define NHEADS  16
#define NKV     4
#define HEADD   128
// N_REP = NHEADS / NKV = 4

typedef __attribute__((ext_vector_type(16))) __bf16 v16bf;
typedef __attribute__((ext_vector_type(8)))  float  v8f;
typedef __attribute__((ext_vector_type(4)))  unsigned int u32x4;
typedef __attribute__((ext_vector_type(8)))  unsigned int u32x8;

// ---------------------------------------------------------------------------
// Helpers
// ---------------------------------------------------------------------------
__device__ __forceinline__ __bf16 f2bf(float x) {
    union { float f; unsigned int u; } c; c.f = x;
    unsigned int r = c.u + 0x7FFFu + ((c.u >> 16) & 1u);   // round-to-nearest-even
    union { unsigned short s; __bf16 b; } o;
    o.s = (unsigned short)(r >> 16);
    return o.b;
}

__device__ __forceinline__ v8f zero8() {
    v8f z;
#pragma unroll
    for (int i = 0; i < 8; ++i) z[i] = 0.0f;
    return z;
}

// Wave-relative LDS byte offset of a shared-memory object (generic pointers to
// LDS carry the offset in the low 32 bits).
__device__ __forceinline__ unsigned lds_off(const void* p) {
    return (unsigned)(unsigned long long)(uintptr_t)p;
}

// Async global->LDS 16-byte copy (gfx1250 ASYNC path, tracked by ASYNCcnt).
__device__ __forceinline__ void async_g2l_b128(unsigned ldsoff, const void* gaddr) {
    asm volatile("global_load_async_to_lds_b128 %0, %1, off"
                 : : "v"(ldsoff), "v"(gaddr) : "memory");
}
__device__ __forceinline__ void wait_async0() {
    asm volatile("s_wait_asynccnt 0x0" ::: "memory");
}

// Tensor Data Mover: 2-D bf16 tile load Global->LDS (ISA ch. 8 descriptor).
//   tile = tile_d0 elems (row, contiguous) x tile_d1 rows
//   tensor dims/stride in 2-byte elements.
__device__ __forceinline__ void tdm_load_2d_bf16(unsigned lds_byte_off,
                                                 const void* gaddr,
                                                 unsigned tile_d0, unsigned tile_d1,
                                                 unsigned tensor_d0, unsigned tensor_d1,
                                                 unsigned long long d0_stride)
{
    const unsigned long long ga = (unsigned long long)(uintptr_t)gaddr;
    u32x4 g0;
    g0[0] = 1u;                                              // count=1 (user D#)
    g0[1] = lds_byte_off;                                    // lds_addr
    g0[2] = (unsigned)(ga & 0xFFFFFFFFu);                    // global_addr[31:0]
    g0[3] = (unsigned)((ga >> 32) & 0x01FFFFFFu) | (2u << 30); // [56:32] + type=2
    u32x8 g1;
    g1[0] = (1u << 16);                                      // data_size=1 (2B), wg_mask=0
    g1[1] = (tensor_d0 & 0xFFFFu) << 16;                     // tensor_dim0[15:0] @ bit48
    g1[2] = (tensor_d0 >> 16) | ((tensor_d1 & 0xFFFFu) << 16); // dim0 hi | dim1 lo
    g1[3] = (tensor_d1 >> 16) | (tile_d0 << 16);             // dim1 hi | tile_dim0
    g1[4] = tile_d1 & 0xFFFFu;                               // tile_dim1 (tile_dim2=0)
    g1[5] = (unsigned)(d0_stride & 0xFFFFFFFFu);             // dim0_stride[31:0]
    g1[6] = (unsigned)((d0_stride >> 32) & 0xFFFFu);         // dim0_stride[47:32]
    g1[7] = 0u;
    asm volatile("tensor_load_to_lds %0, %1" : : "s"(g0), "s"(g1) : "memory");
}

// Load one 16x32 bf16 fragment in the CDNA5 WMMA "A" striping:
//   lanes 0-15: row = lane,    K in {0..7, 16..23}
//   lanes 16-31: row = lane-16, K in {8..15, 24..31}
// The same striping serves B-matrix fragments when `base` points at the
// N-major (i.e. B^T row-major) tile: lane indexes the N column.
__device__ __forceinline__ v16bf load_frag(const __bf16* base, int stride, int lane) {
    const int mr = lane & 15;
    const int kb = (lane >> 4) * 8;
    const __bf16* row = base + mr * stride + kb;
    union { unsigned int u[8]; v16bf v; } f;
    f.u[0] = *(const unsigned int*)(row + 0);
    f.u[1] = *(const unsigned int*)(row + 2);
    f.u[2] = *(const unsigned int*)(row + 4);
    f.u[3] = *(const unsigned int*)(row + 6);
    f.u[4] = *(const unsigned int*)(row + 16);
    f.u[5] = *(const unsigned int*)(row + 18);
    f.u[6] = *(const unsigned int*)(row + 20);
    f.u[7] = *(const unsigned int*)(row + 22);
    return f.v;
}

__device__ __forceinline__ v8f wmma_bf16(v16bf a, v16bf b, v8f c) {
    return __builtin_amdgcn_wmma_f32_16x16x32_bf16(
        /*neg_a=*/false, a, /*neg_b=*/false, b,
        /*c_mod=*/(short)0, c, /*reuse_a=*/false, /*reuse_b=*/false);
}

// ---------------------------------------------------------------------------
// Kernel 1: projection GEMM  out_bf16[b][h][s][hd] = rope?( X @ W )
//   X: [4096 x 2048] f32 (rows = b*S + s), W: [2048 x N] f32
//   block = 128 threads (4 waves), tile 64x64, K-step 32
// ---------------------------------------------------------------------------
__global__ __launch_bounds__(128)
void gemm_proj_kernel(const float* __restrict__ X, const float* __restrict__ W,
                      __bf16* __restrict__ out,
                      const float* __restrict__ fcos, const float* __restrict__ fsin,
                      int N, int nheads, int do_rope)
{
    __shared__ __align__(16) __bf16 As[64][40];
    __shared__ __align__(16) __bf16 Bs[64][40];
    __shared__ __align__(16) float  Cs[64][66];

    const int tid  = threadIdx.x;
    const int lane = tid & 31;
    const int w    = tid >> 5;
    const int half = lane >> 4;
    const int nn   = lane & 15;
    const int m0   = blockIdx.x * 64;
    const int n0   = blockIdx.y * 64;

    v8f acc[4];
#pragma unroll
    for (int t = 0; t < 4; ++t) acc[t] = zero8();

    for (int kb = 0; kb < DMODEL; kb += 32) {
        __syncthreads();
        // --- stage X tile (64x32 f32 -> bf16) ---
        {
            const int rr = tid >> 3;          // 0..15
            const int cc = (tid & 7) * 4;     // 0..28
#pragma unroll
            for (int p = 0; p < 4; ++p) {
                const int row = rr + p * 16;
                const float4 xv = *(const float4*)(X + (size_t)(m0 + row) * DMODEL + kb + cc);
                __bf16* dst = &As[row][cc];
                dst[0] = f2bf(xv.x); dst[1] = f2bf(xv.y);
                dst[2] = f2bf(xv.z); dst[3] = f2bf(xv.w);
            }
        }
        // --- stage W tile transposed: Bs[n][k] = W[kb+k][n0+n] ---
        {
            const int wn = tid & 63;
            const int k2 = tid >> 6;          // 0..1
#pragma unroll
            for (int p = 0; p < 16; ++p) {
                const int kk = k2 + p * 2;
                Bs[wn][kk] = f2bf(W[(size_t)(kb + kk) * N + n0 + wn]);
            }
            if (kb + 32 < DMODEL)             // prefetch next W tile
                __builtin_prefetch(&W[(size_t)(kb + 32 + k2) * N + n0 + wn], 0, 1);
        }
        __syncthreads();
        // --- WMMA: batch all fragment loads, then issue WMMAs back-to-back ---
        const v16bf a = load_frag(&As[w * 16][0], 40, lane);
        v16bf bfrag[4];
#pragma unroll
        for (int nt = 0; nt < 4; ++nt)
            bfrag[nt] = load_frag(&Bs[nt * 16][0], 40, lane);
#pragma unroll
        for (int nt = 0; nt < 4; ++nt)
            acc[nt] = wmma_bf16(a, bfrag[nt], acc[nt]);
    }

    // --- epilogue: stage to LDS, apply RoPE pairs, store bf16 transposed ---
    __syncthreads();
#pragma unroll
    for (int nt = 0; nt < 4; ++nt)
#pragma unroll
        for (int r = 0; r < 8; ++r)
            Cs[w * 16 + r + half * 8][nt * 16 + nn] = acc[nt][r];
    __syncthreads();

    for (int idx = tid; idx < 64 * 32; idx += 128) {
        const int i   = idx >> 5;          // local row
        const int col = (idx & 31) * 2;    // local even column
        float re = Cs[i][col];
        float im = Cs[i][col + 1];
        const int gr = m0 + i;
        const int b  = gr >> 11;           // / SEQ
        const int s  = gr & (SEQ - 1);
        const int gc = n0 + col;
        const int h  = gc >> 7;            // / HEADD
        const int hd = gc & (HEADD - 1);
        float o0 = re, o1 = im;
        if (do_rope) {
            const int p = hd >> 1;
            const float c = fcos[s * (HEADD / 2) + p];
            const float sn = fsin[s * (HEADD / 2) + p];
            o0 = re * c - im * sn;
            o1 = re * sn + im * c;
        }
        const size_t base = ((size_t)(b * nheads + h) * SEQ + s) * HEADD + hd;
        out[base]     = f2bf(o0);
        out[base + 1] = f2bf(o1);
    }
}

// ---------------------------------------------------------------------------
// Kernel 2: flash attention (causal, GQA), online softmax.
//   grid = (SEQ/64, BATCH*NHEADS), block = 128 (4 waves)
//   K tiles double-buffered through the Tensor Data Mover (TENSORcnt pipelined);
//   V tile transposed manually.
// ---------------------------------------------------------------------------
__global__ __launch_bounds__(128)
void attn_kernel(const __bf16* __restrict__ qb_, const __bf16* __restrict__ kb_,
                 const __bf16* __restrict__ vb_, __bf16* __restrict__ aout)
{
    __shared__ __align__(16) __bf16 Ks[2][64][HEADD];  // double-buffered TDM target
    __shared__ __align__(16) __bf16 Vst[HEADD][72];    // V transposed: [hd][key]
    __shared__ __align__(16) float  Ss[64][68];        // raw scores
    __shared__ __align__(16) __bf16 Ps[64][72];        // probs (bf16)
    __shared__ float mrow[64], lrow[64], crow[64];

    const int tid  = threadIdx.x;
    const int lane = tid & 31;
    const int w    = tid >> 5;
    const int half = lane >> 4;
    const int nn   = lane & 15;
    const int qb   = blockIdx.x;
    const int bh   = blockIdx.y;
    const int b    = bh >> 4;
    const int h    = bh & 15;
    const int kvh  = h >> 2;                 // N_REP = 4

    const __bf16* qptr = qb_ + (size_t)(b * NHEADS + h) * SEQ * HEADD;
    const __bf16* kptr = kb_ + (size_t)(b * NKV + kvh) * SEQ * HEADD;
    const __bf16* vptr = vb_ + (size_t)(b * NKV + kvh) * SEQ * HEADD;
    const unsigned ksoff[2] = { lds_off(&Ks[0][0][0]), lds_off(&Ks[1][0][0]) };

    // Q fragments live in registers for the whole kernel (4 K-steps over HD=128)
    v16bf qf[4];
    {
        const __bf16* qrow = qptr + (size_t)(qb * 64 + w * 16) * HEADD;
#pragma unroll
        for (int kk = 0; kk < 4; ++kk)
            qf[kk] = load_frag(qrow + kk * 32, HEADD, lane);
    }

    v8f oacc[8];
#pragma unroll
    for (int t = 0; t < 8; ++t) oacc[t] = zero8();

    if (tid < 64) { mrow[tid] = -3.0e38f; lrow[tid] = 0.0f; }

    // Preload first K tile via TDM (wave 0 only).
    if (tid < 32)
        tdm_load_2d_bf16(ksoff[0], kptr, HEADD, 64, HEADD, SEQ, HEADD);

    const float scale = 0.08838834764831845f;   // 1/sqrt(128)

    for (int kbk = 0; kbk <= qb; ++kbk) {
        const int cur = kbk & 1;
        __syncthreads();   // all waves done with both K buffers from prior iters
        // --- prefetch next K tile into the alternate buffer, then wait until
        //     only that prefetch is outstanding (=> current tile complete) ---
        if (tid < 32) {
            if (kbk + 1 <= qb) {
                tdm_load_2d_bf16(ksoff[cur ^ 1],
                                 kptr + (size_t)((kbk + 1) * 64) * HEADD,
                                 HEADD, 64, HEADD, SEQ, HEADD);
                __builtin_amdgcn_s_wait_tensorcnt((unsigned short)1);
            } else {
                __builtin_amdgcn_s_wait_tensorcnt((unsigned short)0);
            }
        }
        // --- V tile: load row-major, store transposed into LDS ---
#pragma unroll
        for (int p = 0; p < 8; ++p) {
            const int linear = p * 128 + tid;     // 0..1023
            const int row = linear >> 4;          // key local 0..63
            const int c8  = (linear & 15) * 8;    // hd base
            const uint4 vv4 = *(const uint4*)(vptr + (size_t)(kbk * 64 + row) * HEADD + c8);
            __bf16 tmp[8];
            *(uint4*)tmp = vv4;
#pragma unroll
            for (int e = 0; e < 8; ++e) Vst[c8 + e][row] = tmp[e];
        }
        __syncthreads();

        // --- S = Q @ K^T : batch B-fragment loads, then WMMAs back-to-back ---
        v8f sacc[4];
#pragma unroll
        for (int nt = 0; nt < 4; ++nt) sacc[nt] = zero8();
#pragma unroll
        for (int kk = 0; kk < 4; ++kk) {
            v16bf bfrag[4];
#pragma unroll
            for (int nt = 0; nt < 4; ++nt)
                bfrag[nt] = load_frag(&Ks[cur][nt * 16][kk * 32], HEADD, lane);
#pragma unroll
            for (int nt = 0; nt < 4; ++nt)
                sacc[nt] = wmma_bf16(qf[kk], bfrag[nt], sacc[nt]);
        }
        // --- scale + causal mask + stage scores to LDS ---
#pragma unroll
        for (int nt = 0; nt < 4; ++nt) {
            const int kg = kbk * 64 + nt * 16 + nn;
#pragma unroll
            for (int r = 0; r < 8; ++r) {
                const int qg = qb * 64 + w * 16 + r + half * 8;
                float sv = sacc[nt][r] * scale;
                if (kg > qg) sv = -3.0e38f;
                Ss[w * 16 + r + half * 8][nt * 16 + nn] = sv;
            }
        }
        __syncthreads();

        // --- online softmax, one thread per q row ---
        if (tid < 64) {
            const int i = tid;
            const float mo = mrow[i];
            float mx = mo;
            for (int j = 0; j < 64; ++j) mx = fmaxf(mx, Ss[i][j]);
            const float corr = __expf(mo - mx);
            float sum = 0.0f;
            for (int j = 0; j < 64; ++j) {
                const float pv = __expf(Ss[i][j] - mx);
                Ps[i][j] = f2bf(pv);
                sum += pv;
            }
            lrow[i] = lrow[i] * corr + sum;
            mrow[i] = mx;
            crow[i] = corr;
        }
        __syncthreads();

        // --- rescale O, then O += P @ V ---
        float cf[8];
#pragma unroll
        for (int r = 0; r < 8; ++r) cf[r] = crow[w * 16 + r + half * 8];
#pragma unroll
        for (int t = 0; t < 8; ++t)
#pragma unroll
            for (int r = 0; r < 8; ++r) oacc[t][r] *= cf[r];

        v16bf pf0 = load_frag(&Ps[w * 16][0],  72, lane);
        v16bf pf1 = load_frag(&Ps[w * 16][32], 72, lane);
#pragma unroll
        for (int t = 0; t < 8; ++t) {
            const v16bf b0 = load_frag(&Vst[t * 16][0],  72, lane);
            const v16bf b1 = load_frag(&Vst[t * 16][32], 72, lane);
            oacc[t] = wmma_bf16(pf0, b0, oacc[t]);
            oacc[t] = wmma_bf16(pf1, b1, oacc[t]);
        }
    }

    // --- normalize and store bf16 attn output [b][s][h*HD+hd] ---
    float invl[8];
#pragma unroll
    for (int r = 0; r < 8; ++r) invl[r] = 1.0f / lrow[w * 16 + r + half * 8];
#pragma unroll
    for (int t = 0; t < 8; ++t) {
#pragma unroll
        for (int r = 0; r < 8; ++r) {
            const int s = qb * 64 + w * 16 + r + half * 8;
            const size_t addr = ((size_t)(b * SEQ) + s) * (NHEADS * HEADD)
                              + h * HEADD + t * 16 + nn;
            aout[addr] = f2bf(oacc[t][r] * invl[r]);
        }
    }
}

// ---------------------------------------------------------------------------
// Kernel 3: output GEMM  out_f32[4096 x 2048] = A_bf16 @ wo_f32
//   A tile staged with async global->LDS copies (ASYNCcnt fenced).
// ---------------------------------------------------------------------------
__global__ __launch_bounds__(128)
void gemm_out_kernel(const __bf16* __restrict__ A, const float* __restrict__ W,
                     float* __restrict__ out)
{
    __shared__ __align__(16) __bf16 As[64][40];
    __shared__ __align__(16) __bf16 Bs[64][40];

    const int tid  = threadIdx.x;
    const int lane = tid & 31;
    const int w    = tid >> 5;
    const int half = lane >> 4;
    const int nn   = lane & 15;
    const int m0   = blockIdx.x * 64;
    const int n0   = blockIdx.y * 64;
    const unsigned asoff = lds_off(&As[0][0]);

    v8f acc[4];
#pragma unroll
    for (int t = 0; t < 4; ++t) acc[t] = zero8();

    for (int kb = 0; kb < DMODEL; kb += 32) {
        __syncthreads();
        // stage A tile (already bf16) via async copies: 64 rows x 32 elems
        //   = 256 x 16B transfers; lanes supply LDS offset + global address
#pragma unroll
        for (int p = 0; p < 2; ++p) {
            const int linear = p * 128 + tid;   // 0..255
            const int row = linear >> 2;        // 0..63
            const int c8  = (linear & 3) * 8;   // 0,8,16,24
            async_g2l_b128(asoff + (unsigned)(row * 80 + c8 * 2),
                           A + (size_t)(m0 + row) * DMODEL + kb + c8);
        }
        // stage wo tile transposed (f32 -> bf16) while async copies fly
        {
            const int wn = tid & 63;
            const int k2 = tid >> 6;
#pragma unroll
            for (int p = 0; p < 16; ++p) {
                const int kk = k2 + p * 2;
                Bs[wn][kk] = f2bf(W[(size_t)(kb + kk) * DMODEL + n0 + wn]);
            }
            if (kb + 32 < DMODEL)               // prefetch next W tile
                __builtin_prefetch(&W[(size_t)(kb + 32 + k2) * DMODEL + n0 + wn], 0, 1);
        }
        wait_async0();
        __syncthreads();
        const v16bf a = load_frag(&As[w * 16][0], 40, lane);
        v16bf bfrag[4];
#pragma unroll
        for (int nt = 0; nt < 4; ++nt)
            bfrag[nt] = load_frag(&Bs[nt * 16][0], 40, lane);
#pragma unroll
        for (int nt = 0; nt < 4; ++nt)
            acc[nt] = wmma_bf16(a, bfrag[nt], acc[nt]);
    }

    // direct f32 stores
#pragma unroll
    for (int nt = 0; nt < 4; ++nt)
#pragma unroll
        for (int r = 0; r < 8; ++r)
            out[(size_t)(m0 + w * 16 + r + half * 8) * DMODEL + n0 + nt * 16 + nn]
                = acc[nt][r];
}

// ---------------------------------------------------------------------------
// Launch
// ---------------------------------------------------------------------------
extern "C" void kernel_launch(void* const* d_in, const int* in_sizes, int n_in,
                              void* d_out, int out_size, void* d_ws, size_t ws_size,
                              hipStream_t stream) {
    const float* x    = (const float*)d_in[0];
    const float* wq   = (const float*)d_in[1];
    const float* wk   = (const float*)d_in[2];
    const float* wv   = (const float*)d_in[3];
    const float* wo   = (const float*)d_in[4];
    const float* fcos = (const float*)d_in[5];
    const float* fsin = (const float*)d_in[6];
    float* out = (float*)d_out;

    char* ws = (char*)d_ws;
    __bf16* qbuf = (__bf16*)(ws);                                  // 16 MB
    __bf16* kbuf = (__bf16*)(ws + (size_t)16 * 1024 * 1024);       //  4 MB
    __bf16* vbuf = (__bf16*)(ws + (size_t)20 * 1024 * 1024);       //  4 MB
    __bf16* abuf = (__bf16*)(ws + (size_t)24 * 1024 * 1024);       // 16 MB

    const dim3 blk(128);
    // Q/K/V projections (+RoPE on q,k)
    gemm_proj_kernel<<<dim3(64, 32), blk, 0, stream>>>(x, wq, qbuf, fcos, fsin,
                                                       NHEADS * HEADD, NHEADS, 1);
    gemm_proj_kernel<<<dim3(64, 8),  blk, 0, stream>>>(x, wk, kbuf, fcos, fsin,
                                                       NKV * HEADD, NKV, 1);
    gemm_proj_kernel<<<dim3(64, 8),  blk, 0, stream>>>(x, wv, vbuf, fcos, fsin,
                                                       NKV * HEADD, NKV, 0);
    // causal GQA flash attention
    attn_kernel<<<dim3(SEQ / 64, BATCH * NHEADS), blk, 0, stream>>>(qbuf, kbuf, vbuf, abuf);
    // output projection
    gemm_out_kernel<<<dim3(64, 32), blk, 0, stream>>>(abuf, wo, out);
}
